// MambaFeatureEnhancer_36318243455456
// MI455X (gfx1250) — compile-verified, matrix operand order checked by reading, not run on hardware
//
#include <hip/hip_runtime.h>
#include <hip/hip_bf16.h>
#include <math.h>

typedef __bf16 bf16;
typedef __attribute__((ext_vector_type(16))) __bf16 v16bf;
typedef __attribute__((ext_vector_type(8)))  __bf16 v8bf;
typedef __attribute__((ext_vector_type(4)))  __bf16 v4bf;
typedef __attribute__((ext_vector_type(8)))  float  v8f;

#define B_  2
#define C_  128
#define HW_ 4096
#define M_  (B_ * HW_)      // 8192 tokens
#define DI_ 256             // d_inner
#define NS_ 16              // d_state
#define RK_ 8               // dt_rank
#define NLAYER_ 2

// ---------------------------------------------------------------------------
// WMMA GEMM:  D[m,n] = sum_k A[m,k] * W[n,k]   (A: M x K bf16, W: Npad x K bf16)
// block = 256 threads = 8 waves. Block tile: 128 rows x 64 cols, K stepped 64.
// W tile (64x64 bf16 = 8KB) staged into double-buffered LDS with the CDNA5
// async-copy path (global_load_async_to_lds_b128, ASYNCcnt): one barrier per
// 64-K step, next tile in flight while 8 WMMAs consume the current one.
// Epilogue: optional bias, exact GELU, residual add, f32/bf16 stores, or the
// final transposed store fused with the outer residual (x).
// ---------------------------------------------------------------------------
__global__ __launch_bounds__(256)
void gemm_wmma_bf16(const bf16* __restrict__ A, const bf16* __restrict__ W,
                    int M, int K, int ldD,
                    const float* __restrict__ bias,
                    const float* __restrict__ resid,
                    float* __restrict__ outF, bf16* __restrict__ outB,
                    int act,
                    const float* __restrict__ xres, float* __restrict__ outT) {
    __shared__ bf16 lds[2][64 * 64];

    const int tid  = threadIdx.x;
    const int lane = tid & 31;
    const int wave = tid >> 5;
    const int mBase = blockIdx.x * 128 + wave * 16;
    const int nBase = blockIdx.y * 64;

    v8f acc[4];
    #pragma unroll
    for (int i = 0; i < 4; ++i)
        acc[i] = (v8f){0.f,0.f,0.f,0.f,0.f,0.f,0.f,0.f};

    const int row   = mBase + (lane & 15);
    const int khalf = (lane >> 4) * 8;        // 0 or 8 (A/B 16-bit frag layout)
    const int wr = tid >> 2;                  // 0..63  (W tile stage row)
    const int wc = (tid & 3) * 16;            // 0,16,32,48 (halves; 32B/thread)

    // LDS byte offsets for this thread's two 16B staging slots in each buffer
    unsigned ldsOff[2];
    ldsOff[0] = (unsigned)(size_t)&lds[0][wr * 64 + wc];
    ldsOff[1] = (unsigned)(size_t)&lds[1][wr * 64 + wc];
    const size_t wrow = (size_t)(nBase + wr) * K + wc;

    // async copy of one W K-tile (64x64 bf16) directly into LDS buffer `b`
    auto issueTile = [&](int k0, int b) {
        unsigned long long ga = (unsigned long long)(const void*)&W[wrow + k0];
        asm volatile("global_load_async_to_lds_b128 %0, %1, off"
                     :: "v"(ldsOff[b]), "v"(ga) : "memory");
        asm volatile("global_load_async_to_lds_b128 %0, %1, off offset:16"
                     :: "v"(ldsOff[b]), "v"(ga) : "memory");
    };

    issueTile(0, 0);
    int buf = 0;
    for (int k0 = 0; k0 < K; k0 += 64) {
        // tile k0 has landed; everyone is done reading the buffer that the
        // next async copy will overwrite (read in the previous step, pre-barrier)
        asm volatile("s_wait_asynccnt 0x0" ::: "memory");
        __syncthreads();
        if (k0 + 64  < K) issueTile(k0 + 64, buf ^ 1);
        if (k0 + 128 < K) __builtin_prefetch(&A[(size_t)row * K + k0 + 128], 0, 3);

        const bf16* lb = &lds[buf][0];
        #pragma unroll
        for (int ks = 0; ks < 64; ks += 32) {
            // A fragment (16x32 bf16, documented lane layout)
            const bf16* ap = &A[(size_t)row * K + k0 + ks + khalf];
            union { v16bf v; v8bf h[2]; } af;
            af.h[0] = *(const v8bf*)ap;
            af.h[1] = *(const v8bf*)(ap + 16);

            #pragma unroll
            for (int nt = 0; nt < 4; ++nt) {
                const bf16* bp = &lb[(nt * 16 + (lane & 15)) * 64 + ks + khalf];
                union { v16bf v; v8bf h[2]; } bfr;
                bfr.h[0] = *(const v8bf*)bp;
                bfr.h[1] = *(const v8bf*)(bp + 16);
                acc[nt] = __builtin_amdgcn_wmma_f32_16x16x32_bf16(
                    false, af.v, false, bfr.v, (short)0, acc[nt], false, false);
            }
        }
        buf ^= 1;
    }

    // Epilogue. D layout: lanes 0-15 -> M = mBase+v, lanes 16-31 -> M = mBase+8+v.
    const int rbase = mBase + ((lane >> 4) ? 8 : 0);
    const int ncol0 = nBase + (lane & 15);
    #pragma unroll
    for (int nt = 0; nt < 4; ++nt) {
        const int n = ncol0 + nt * 16;
        #pragma unroll
        for (int v = 0; v < 8; ++v) {
            const int r = rbase + v;
            float val = acc[nt][v];
            if (bias)  val += bias[n];
            if (act == 1)  // exact GELU
                val = 0.5f * val * (1.f + erff(val * 0.70710678118654752f));
            if (resid) val += resid[(size_t)r * ldD + n];
            if (outF)  outF[(size_t)r * ldD + n] = val;
            if (outB)  outB[(size_t)r * ldD + n] = (bf16)val;
            if (outT) {
                const int b = r >> 12;            // L = 4096
                const int l = r & 4095;
                const size_t o = ((size_t)(b * C_ + n)) * HW_ + l;
                outT[o] = val + xres[o];
            }
        }
    }
}

// ---------------------------------------------------------------------------
// fp32 -> bf16 conversion with optional zero row padding (rowsPad >= rows)
// ---------------------------------------------------------------------------
__global__ void cvt_pad_bf16(const float* __restrict__ src, bf16* __restrict__ dst,
                             int rows, int cols, long total) {
    long idx = (long)blockIdx.x * blockDim.x + threadIdx.x;
    if (idx >= total) return;
    long r = idx / cols;
    dst[idx] = (r < rows) ? (bf16)src[idx] : (bf16)0.f;
}

// x (B,C,HW) fp32 -> token-major (B*HW, C) bf16
__global__ void xpose_bf16(const float* __restrict__ x, bf16* __restrict__ out) {
    int idx = blockIdx.x * blockDim.x + threadIdx.x;   // < B*HW*C
    int c = idx & (C_ - 1);
    int m = idx >> 7;
    int l = m & (HW_ - 1);
    int b = m >> 12;
    out[idx] = (bf16)x[(((size_t)b * C_ + c) << 12) + l];
}

// ---------------------------------------------------------------------------
// LayerNorm over C=128; one wave per row; output bf16 (feeds next GEMM as A)
// ---------------------------------------------------------------------------
__global__ __launch_bounds__(256)
void ln_bf16(const float* __restrict__ x, const float* __restrict__ g,
             const float* __restrict__ b, bf16* __restrict__ out) {
    const int lane = threadIdx.x & 31;
    const int wave = threadIdx.x >> 5;
    const int row  = blockIdx.x * 8 + wave;
    const float4 v = ((const float4*)(x + (size_t)row * C_))[lane];
    float s  = v.x + v.y + v.z + v.w;
    float sq = v.x * v.x + v.y * v.y + v.z * v.z + v.w * v.w;
    #pragma unroll
    for (int o = 16; o >= 1; o >>= 1) {
        s  += __shfl_xor(s,  o, 32);
        sq += __shfl_xor(sq, o, 32);
    }
    const float mean = s * (1.f / C_);
    const float var  = sq * (1.f / C_) - mean * mean;
    const float rs   = rsqrtf(var + 1e-5f);
    const int cb = lane * 4;
    v4bf o4;
    o4[0] = (bf16)((v.x - mean) * rs * g[cb + 0] + b[cb + 0]);
    o4[1] = (bf16)((v.y - mean) * rs * g[cb + 1] + b[cb + 1]);
    o4[2] = (bf16)((v.z - mean) * rs * g[cb + 2] + b[cb + 2]);
    o4[3] = (bf16)((v.w - mean) * rs * g[cb + 3] + b[cb + 3]);
    *(v4bf*)(out + (size_t)row * C_ + cb) = o4;
}

// ---------------------------------------------------------------------------
// causal depthwise conv1d (k=4, left pad 3) + SiLU; u part of xz (stride 512)
// ---------------------------------------------------------------------------
__global__ void conv_silu(const float* __restrict__ xz, const float* __restrict__ w,
                          const float* __restrict__ cb,
                          float* __restrict__ uf, bf16* __restrict__ ub) {
    int idx = blockIdx.x * blockDim.x + threadIdx.x;    // M_*DI_
    if (idx >= M_ * DI_) return;
    const int d = idx & (DI_ - 1);
    const int m = idx >> 8;
    const int l = m & (HW_ - 1);
    float a = cb[d];
    #pragma unroll
    for (int j = 0; j < 4; ++j) {
        const int lj = l - 3 + j;
        if (lj >= 0) a += xz[(size_t)(m - 3 + j) * 512 + d] * w[d * 4 + j];
    }
    const float s = a / (1.f + expf(-a));
    uf[idx] = s;
    ub[idx] = (bf16)s;
}

// dt = softplus(dbl[:, :8] @ dt_w.T + dt_b)   (K=8 -> plain VALU)
__global__ void dtproj_softplus(const float* __restrict__ dbl,
                                const float* __restrict__ dw,
                                const float* __restrict__ db,
                                float* __restrict__ dt) {
    int idx = blockIdx.x * blockDim.x + threadIdx.x;    // M_*DI_
    if (idx >= M_ * DI_) return;
    const int d = idx & (DI_ - 1);
    const int m = idx >> 8;
    const float* dr = dbl + (size_t)m * 64;
    float a = db[d];
    #pragma unroll
    for (int r = 0; r < RK_; ++r) a += dr[r] * dw[d * RK_ + r];
    dt[idx] = (a > 20.f) ? a : log1pf(expf(a));
}

// ---------------------------------------------------------------------------
// Selective scan: one lane per (b, d, n); sequential over L.
// Shuffle-reduce over n (16 lanes), fused  (+u*D) * silu(z)  gate, bf16 out.
// Unrolled so the independent per-step loads run ahead of the recurrence chain.
// ---------------------------------------------------------------------------
__global__ __launch_bounds__(256)
void ssm_scan(const float* __restrict__ dt, const float* __restrict__ u,
              const float* __restrict__ dbl, const float* __restrict__ xz,
              const float* __restrict__ A_log, const float* __restrict__ ssmD,
              bf16* __restrict__ yg) {
    const int idx = blockIdx.x * blockDim.x + threadIdx.x; // B_*DI_*NS_ = 8192
    const int n = idx & (NS_ - 1);
    const int d = (idx >> 4) & (DI_ - 1);
    const int b = idx >> 12;
    const float a  = -expf(A_log[d * NS_ + n]);
    const float Dd = ssmD[d];
    float h = 0.f;
    #pragma unroll 4
    for (int l = 0; l < HW_; ++l) {
        const size_t m  = (size_t)b * HW_ + l;
        const float dtv = dt[m * DI_ + d];
        const float uv  = u[m * DI_ + d];
        const float Bv  = dbl[m * 64 + RK_ + n];
        const float Cv  = dbl[m * 64 + RK_ + NS_ + n];
        h = expf(dtv * a) * h + dtv * uv * Bv;
        float yp = h * Cv;
        yp += __shfl_xor(yp, 1, 16);
        yp += __shfl_xor(yp, 2, 16);
        yp += __shfl_xor(yp, 4, 16);
        yp += __shfl_xor(yp, 8, 16);
        if (n == 0) {
            const float y = yp + uv * Dd;
            const float z = xz[m * 512 + DI_ + d];
            yg[m * DI_ + d] = (bf16)(y * (z / (1.f + expf(-z))));
        }
    }
}

// ---------------------------------------------------------------------------
extern "C" void kernel_launch(void* const* d_in, const int* in_sizes, int n_in,
                              void* d_out, int out_size, void* d_ws, size_t ws_size,
                              hipStream_t stream) {
    const float* x       = (const float*)d_in[0];
    const float* embed_w = (const float*)d_in[1];
    const float* embed_b = (const float*)d_in[2];
    const float* outc_w  = (const float*)d_in[3];
    const float* outc_b  = (const float*)d_in[4];
    const float* ln_g    = (const float*)d_in[5];
    const float* ln_bp   = (const float*)d_in[6];
    const float* in_w    = (const float*)d_in[7];
    const float* conv_w  = (const float*)d_in[8];
    const float* conv_b  = (const float*)d_in[9];
    const float* xproj_w = (const float*)d_in[10];
    const float* dt_w    = (const float*)d_in[11];
    const float* dt_b    = (const float*)d_in[12];
    const float* A_log   = (const float*)d_in[13];
    const float* ssm_D   = (const float*)d_in[14];
    const float* mout_w  = (const float*)d_in[15];
    const float* ffn_w1  = (const float*)d_in[16];
    const float* ffn_b1  = (const float*)d_in[17];
    const float* ffn_w2  = (const float*)d_in[18];
    const float* ffn_b2  = (const float*)d_in[19];
    (void)in_sizes; (void)n_in; (void)out_size; (void)ws_size;

    char* wp = (char*)d_ws;
    auto alloc = [&](size_t bytes) -> void* {
        void* p = (void*)wp;
        wp += (bytes + 255) & ~(size_t)255;
        return p;
    };
    const int M = M_;
    float* t      = (float*)alloc((size_t)M * 128 * 4);
    bf16*  abf    = (bf16*) alloc((size_t)M * 128 * 2);
    float* xz     = (float*)alloc((size_t)M * 512 * 4);
    float* uf     = (float*)alloc((size_t)M * 256 * 4);
    bf16*  ubf    = (bf16*) alloc((size_t)M * 256 * 2);
    float* dbl    = (float*)alloc((size_t)M * 64  * 4);
    float* dtb    = (float*)alloc((size_t)M * 256 * 4);
    bf16*  ygbf   = (bf16*) alloc((size_t)M * 256 * 2);
    bf16*  midbf  = (bf16*) alloc((size_t)M * 512 * 2);
    bf16*  embw   = (bf16*) alloc(128 * 128 * 2);
    bf16*  outw   = (bf16*) alloc(128 * 128 * 2);
    bf16*  inw    = (bf16*) alloc((size_t)NLAYER_ * 512 * 128 * 2);
    bf16*  xpw    = (bf16*) alloc((size_t)NLAYER_ * 64  * 256 * 2);  // padded 40->64
    bf16*  mow    = (bf16*) alloc((size_t)NLAYER_ * 128 * 256 * 2);
    bf16*  f1w    = (bf16*) alloc((size_t)NLAYER_ * 512 * 128 * 2);
    bf16*  f2w    = (bf16*) alloc((size_t)NLAYER_ * 128 * 512 * 2);

    auto cvt = [&](const float* s, bf16* d, int rows, int rowsPad, int cols) {
        long total = (long)rowsPad * cols;
        cvt_pad_bf16<<<(int)((total + 255) / 256), 256, 0, stream>>>(s, d, rows, cols, total);
    };
    auto gemm = [&](const bf16* A, const bf16* W, int K, int Npad,
                    const float* bias, const float* resid,
                    float* outF, bf16* outB, int act,
                    const float* xres, float* outT) {
        dim3 g(M / 128, Npad / 64), blk(256);
        gemm_wmma_bf16<<<g, blk, 0, stream>>>(A, W, M, K, Npad, bias, resid,
                                              outF, outB, act, xres, outT);
    };

    // weight conversion (once per launch; deterministic)
    cvt(embed_w, embw, 128, 128, 128);
    cvt(outc_w,  outw, 128, 128, 128);
    cvt(in_w,    inw,  NLAYER_ * 512, NLAYER_ * 512, 128);
    for (int i = 0; i < NLAYER_; ++i)
        cvt(xproj_w + (size_t)i * 40 * 256, xpw + (size_t)i * 64 * 256, 40, 64, 256);
    cvt(mout_w,  mow,  NLAYER_ * 128, NLAYER_ * 128, 256);
    cvt(ffn_w1,  f1w,  NLAYER_ * 512, NLAYER_ * 512, 128);
    cvt(ffn_w2,  f2w,  NLAYER_ * 128, NLAYER_ * 128, 512);

    // embed: x -> token-major bf16 -> GEMM -> t (M x 128)
    {
        int total = M * 128;
        xpose_bf16<<<total / 256, 256, 0, stream>>>(x, abf);
        gemm(abf, embw, 128, 128, embed_b, nullptr, t, nullptr, 0, nullptr, nullptr);
    }

    for (int i = 0; i < NLAYER_; ++i) {
        const float* g = ln_g  + (size_t)i * 128;
        const float* b = ln_bp + (size_t)i * 128;

        // --- mamba block ---
        ln_bf16<<<M / 8, 256, 0, stream>>>(t, g, b, abf);
        gemm(abf, inw + (size_t)i * 512 * 128, 128, 512,
             nullptr, nullptr, xz, nullptr, 0, nullptr, nullptr);
        conv_silu<<<(M * 256) / 256, 256, 0, stream>>>(
            xz, conv_w + (size_t)i * 256 * 4, conv_b + (size_t)i * 256, uf, ubf);
        gemm(ubf, xpw + (size_t)i * 64 * 256, 256, 64,
             nullptr, nullptr, dbl, nullptr, 0, nullptr, nullptr);
        dtproj_softplus<<<(M * 256) / 256, 256, 0, stream>>>(
            dbl, dt_w + (size_t)i * 256 * 8, dt_b + (size_t)i * 256, dtb);
        ssm_scan<<<(B_ * DI_ * NS_) / 256, 256, 0, stream>>>(
            dtb, uf, dbl, xz, A_log + (size_t)i * 256 * 16,
            ssm_D + (size_t)i * 256, ygbf);
        gemm(ygbf, mow + (size_t)i * 128 * 256, 256, 128,
             nullptr, t, t, nullptr, 0, nullptr, nullptr);   // + residual, in-place

        // --- FFN block (same LN params, per reference) ---
        ln_bf16<<<M / 8, 256, 0, stream>>>(t, g, b, abf);
        gemm(abf, f1w + (size_t)i * 512 * 128, 128, 512,
             ffn_b1 + (size_t)i * 512, nullptr, nullptr, midbf, 1, nullptr, nullptr);
        gemm(midbf, f2w + (size_t)i * 128 * 512, 512, 128,
             ffn_b2 + (size_t)i * 128, t, t, nullptr, 0, nullptr, nullptr);
    }

    // out conv + outer residual, fused transposed store to (B,C,H,W)
    cvt(t, abf, M, M, 128);
    gemm(abf, outw, 128, 128, outc_b, nullptr, nullptr, nullptr, 0,
         x, (float*)d_out);
}
